// BetweennessRoPE_1992864825908
// MI455X (gfx1250) — matrix operand order repaired and checked by reading, not compiled, not run on hardware
//
#include <hip/hip_runtime.h>

// CDNA5 / gfx1250 fused Betweenness-RoPE kernel.
// wave32, one wave per 16-token tile of one (b,h) sequence.

typedef __attribute__((ext_vector_type(16))) __bf16 v16bf;
typedef __attribute__((ext_vector_type(8)))  float  v8f;
typedef __attribute__((ext_vector_type(4)))  int    v4i;

#define AS1 __attribute__((address_space(1)))
#define AS3 __attribute__((address_space(3)))

#define LROW 68              // padded LDS row stride (floats): 16B aligned, bank-spread
#define BDIM 8
#define LDIM 2048
#define HDIM 16
#define DDIM 64
#define TILE 16
#define TILES (LDIM / TILE)  // 128
#define WFRAG_BYTES (32 * 128 * 2)   // 32 lanes x 128 bf16

#if __has_builtin(__builtin_amdgcn_global_load_async_to_lds_b128) && \
    __has_builtin(__builtin_amdgcn_s_wait_asynccnt)
#define USE_ASYNC_LDS 1
#else
#define USE_ASYNC_LDS 0
#endif

// ---- Prep: per-lane pre-swizzled bf16 B-fragments of W^T (B[k][n] = W[e][d]) ----
// 16-bit B layout used here (contiguous K): lanes 0-15 hold K=0..15 (2 per VGPR), lanes 16-31 K=16..31.
__global__ __launch_bounds__(32)
void prep_w_frags(const float* __restrict__ W, __bf16* __restrict__ wfrag)
{
    const int lane = threadIdx.x;
    const int half = lane >> 4;
    const int nq   = lane & 15;
    __bf16* dst = wfrag + lane * 128;
    #pragma unroll
    for (int nt = 0; nt < 4; ++nt) {
        const int e = nt * 16 + nq;
        const float* wrow = W + e * DDIM;
        #pragma unroll
        for (int kc = 0; kc < 2; ++kc) {
            const int dbase = kc * 32 + half * 16;
            #pragma unroll
            for (int t = 0; t < 8; ++t) {
                float2 w2 = *(const float2*)(wrow + dbase + 2 * t);
                dst[(nt * 2 + kc) * 16 + 2 * t]     = (__bf16)w2.x;
                dst[(nt * 2 + kc) * 16 + 2 * t + 1] = (__bf16)w2.y;
            }
        }
    }
}

__global__ __launch_bounds__(32)
void betweenness_rope_fused(const float* __restrict__ x,
                            const float* __restrict__ W,
                            const __bf16* __restrict__ wfrag,   // may be null
                            const float* __restrict__ gate,
                            float* __restrict__ out,
                            int useFrag)
{
    __shared__ float xs[18 * LROW];   // 18 rows: tile + halo (edge-clamped)
    __shared__ float adjLDS[TILE];

    const int blk  = blockIdx.x;
    const int tile = blk % TILES;
    const int h    = (blk / TILES) % HDIM;
    const int b    = blk / (TILES * HDIM);
    const int l0   = tile * TILE;
    const int lane = threadIdx.x;
    const int half = lane >> 4;      // 0: lanes 0-15, 1: lanes 16-31
    const int nq   = lane & 15;

    // x[b, l, h, :] base; l-stride = H*D = 1024 floats
    const float* xbh = x + ((size_t)b * LDIM * HDIM + h) * DDIM;

    // ---- Stage 18 rows (l0-1 .. l0+16, clamped) into LDS ----
    #pragma unroll
    for (int r2 = 0; r2 < 18; r2 += 2) {
        int r  = r2 + half;                       // two rows per pass
        int c4 = nq * 4;
        int l  = min(max(l0 - 1 + r, 0), LDIM - 1);
        const float* src = xbh + (size_t)l * (HDIM * DDIM) + c4;
#if USE_ASYNC_LDS
        __builtin_amdgcn_global_load_async_to_lds_b128(
            (AS1 v4i*)(src),
            (AS3 v4i*)(&xs[r * LROW + c4]),
            0, 0);
#else
        float4 v = *(const float4*)src;
        *(float4*)(&xs[r * LROW + c4]) = v;
#endif
    }

    // ---- B fragments: prefer pre-swizzled bf16 image from workspace ----
    v16bf bfrag[4][2];
    if (useFrag) {
        const __bf16* src = wfrag + lane * 128;
        #pragma unroll
        for (int nt = 0; nt < 4; ++nt)
            #pragma unroll
            for (int kc = 0; kc < 2; ++kc)
                bfrag[nt][kc] = *(const v16bf*)(src + (nt * 2 + kc) * 16);
    } else {
        #pragma unroll
        for (int nt = 0; nt < 4; ++nt) {
            const int e = nt * 16 + nq;
            const float* wrow = W + e * DDIM;
            #pragma unroll
            for (int kc = 0; kc < 2; ++kc) {
                const int dbase = kc * 32 + half * 16;
                v16bf bb;
                #pragma unroll
                for (int t = 0; t < 8; ++t) {
                    float2 w2 = *(const float2*)(wrow + dbase + 2 * t);
                    bb[2 * t]     = (__bf16)w2.x;
                    bb[2 * t + 1] = (__bf16)w2.y;
                }
                bfrag[nt][kc] = bb;
            }
        }
    }

#if USE_ASYNC_LDS
    __builtin_amdgcn_s_wait_asynccnt(0);
#endif
    __syncthreads();

    // ---- A fragments: row m diffs (bias cancels).  da = c[l]-c[l-1], db = c[l+1]-c[l].
    // 16-bit A 16x32 layout (matches B): lane m<16, VGPR t: K = 16*half + 2t.
    v16bf da[2], db[2];
    #pragma unroll
    for (int kc = 0; kc < 2; ++kc) {
        const float* r0 = &xs[(nq + 0) * LROW];
        const float* r1 = &xs[(nq + 1) * LROW];
        const float* r2 = &xs[(nq + 2) * LROW];
        #pragma unroll
        for (int t = 0; t < 8; ++t) {
            int k = kc * 32 + 16 * half + 2 * t;
            float2 p0 = *(const float2*)(r0 + k);
            float2 p1 = *(const float2*)(r1 + k);
            float2 p2 = *(const float2*)(r2 + k);
            da[kc][2 * t]     = (__bf16)(p1.x - p0.x);
            da[kc][2 * t + 1] = (__bf16)(p1.y - p0.y);
            db[kc][2 * t]     = (__bf16)(p2.x - p1.x);
            db[kc][2 * t + 1] = (__bf16)(p2.y - p1.y);
        }
    }

    // ---- Two M16/N64/K64 GEMMs; d2 tile = Da + Db by linearity (3rd GEMM eliminated).
    float accA[8], accB[8], accV[8];
    #pragma unroll
    for (int r = 0; r < 8; ++r) { accA[r] = 0.f; accB[r] = 0.f; accV[r] = 0.f; }

    #pragma unroll
    for (int nt = 0; nt < 4; ++nt) {
        v8f ca = {}, cb = {};
        ca = __builtin_amdgcn_wmma_f32_16x16x32_bf16(
                 false, da[0], false, bfrag[nt][0], (short)0, ca, false, false);
        ca = __builtin_amdgcn_wmma_f32_16x16x32_bf16(
                 false, da[1], false, bfrag[nt][1], (short)0, ca, false, false);
        cb = __builtin_amdgcn_wmma_f32_16x16x32_bf16(
                 false, db[0], false, bfrag[nt][0], (short)0, cb, false, false);
        cb = __builtin_amdgcn_wmma_f32_16x16x32_bf16(
                 false, db[1], false, bfrag[nt][1], (short)0, cb, false, false);
        #pragma unroll
        for (int r = 0; r < 8; ++r) {
            float va = ca[r], vb = cb[r], vv = va + vb;
            accA[r] += va * va;
            accB[r] += vb * vb;
            accV[r] += vv * vv;
        }
    }

    // ---- Reduce over N (cols) across the 16 lanes of each half (wave32 shuffles) ----
    #pragma unroll
    for (int off = 1; off < 16; off <<= 1) {
        #pragma unroll
        for (int r = 0; r < 8; ++r) {
            accA[r] += __shfl_xor(accA[r], off, 32);
            accB[r] += __shfl_xor(accB[r], off, 32);
            accV[r] += __shfl_xor(accV[r], off, 32);
        }
    }

    // ---- score -> betweenness -> adjust (rows i = 8*half + r) ----
    const float g0 = gate[0];
    if (nq == 0) {
        #pragma unroll
        for (int r = 0; r < 8; ++r) {
            int   i    = half * 8 + r;
            int   l    = l0 + i;
            float d1a  = sqrtf(accA[r]);
            float d1b  = sqrtf(accB[r]);
            float d2   = sqrtf(accV[r]);
            float sc   = fmaxf(0.0f, 1.0f - (d1a + d1b - d2) / fmaxf(d2, 1e-6f));
            float bw   = (l == 0 || l == LDIM - 1) ? 0.0f : sc * (1.0f / (LDIM - 2));
            adjLDS[i]  = g0 * (bw - 0.5f) * 0.1f;
        }
    }
    __syncthreads();

    // ---- RoPE with interpolated tables: j = lane (freq pair), p = 0..15 ----
    // base_freq[j] = 10000^(-2j/64) = 2^(-j*log2(10000)/32).
    // |adjust| < 0.05  =>  lo in {l0+p-1, l0+p}: all angles lie on the integer
    // lattice (l0+p)*fj. Seed one precise sincos, then advance by a 4-FMA
    // rotation per step instead of calling sincos 16 more times.
    const float fj = exp2f(-(float)lane * (13.287712379549449f / 32.0f));
    float sfj, cfj;
    sincosf(fj, &sfj, &cfj);             // rotation constants for +fj
    float c_p, s_p;                      // cos/sin((l0+p)*fj), p = 0 at start
    sincosf((float)l0 * fj, &s_p, &c_p);
    float c_pm1 = c_p * cfj + s_p * sfj; // rotate by -fj -> (l0-1)*fj
    float s_pm1 = s_p * cfj - c_p * sfj;

    float* obase = out + (((size_t)b * LDIM + l0) * HDIM + h) * DDIM;

    #pragma unroll 4
    for (int p = 0; p < TILE; ++p) {
        float fl   = (float)(l0 + p);
        float adj  = adjLDS[p];
        float apos = fminf(fmaxf(fl + adj, 0.0f), (float)(LDIM - 1));
        float lo   = floorf(apos);
        float frac = apos - lo;
        // next lattice point (l0+p+1)*fj via rotation (also serves as "hi" value)
        float c_n = c_p * cfj - s_p * sfj;
        float s_n = s_p * cfj + c_p * sfj;
        bool  atP = (lo == fl);          // exact integer-float compare
        float cl  = atP ? c_p : c_pm1;
        float sl  = atP ? s_p : s_pm1;
        float ch  = atP ? c_n : c_p;
        float sh  = atP ? s_n : s_p;
        float ci  = (1.0f - frac) * cl + frac * ch;
        float si  = (1.0f - frac) * sl + frac * sh;
        float2 xv = *(const float2*)(&xs[(p + 1) * LROW + 2 * lane]);
        float2 ov;
        ov.x = xv.x * ci - xv.y * si;    // out_even
        ov.y = xv.y * ci + xv.x * si;    // out_odd
        *(float2*)(obase + (size_t)p * (HDIM * DDIM) + 2 * lane) = ov;
        c_pm1 = c_p; s_pm1 = s_p;        // advance recurrence
        c_p   = c_n; s_p   = s_n;
    }
}

extern "C" void kernel_launch(void* const* d_in, const int* in_sizes, int n_in,
                              void* d_out, int out_size, void* d_ws, size_t ws_size,
                              hipStream_t stream) {
    (void)in_sizes; (void)n_in; (void)out_size;
    const float* x    = (const float*)d_in[0];
    const float* W    = (const float*)d_in[1];
    // d_in[2] = bias: cancels in all content differences -> unused.
    const float* gate = (const float*)d_in[3];
    float* out = (float*)d_out;

    const int useFrag = (ws_size >= (size_t)WFRAG_BYTES) ? 1 : 0;
    __bf16* wfrag = (__bf16*)d_ws;
    if (useFrag) {
        prep_w_frags<<<1, 32, 0, stream>>>(W, wfrag);
    }

    dim3 grid(BDIM * HDIM * TILES);   // 8*16*128 = 16384 tiles
    dim3 block(32);                   // one wave32 per tile
    betweenness_rope_fused<<<grid, block, 0, stream>>>(x, W, wfrag, gate, out, useFrag);
}
